// SwinBlock_47622597378109
// MI455X (gfx1250) — compile-verified, matrix operand order checked by reading, not run on hardware
//
#include <hip/hip_runtime.h>
#include <math.h>

#define DIMC 192
#define HEADS 6
#define DH 32
#define WINSZ 7
#define NWIN 1024          // 16 batches * 8*8 windows
#define NTOK 64            // padded tokens per window
#define NREAL 49
#define HW 3136
#define HID 768
#define IMG 56

typedef __attribute__((ext_vector_type(16))) _Float16 v16h;
typedef __attribute__((ext_vector_type(8)))  _Float16 v8h;
typedef __attribute__((ext_vector_type(8)))  float    v8f;
typedef int v4i __attribute__((vector_size(16)));      // matches builtin's param type
typedef __attribute__((address_space(1))) v4i* gv4i_p;
typedef __attribute__((address_space(3))) v4i* lv4i_p;

// ---- CDNA5 async global->LDS staging (guarded; falls back to b128 copies) ----
#if defined(__HIP_DEVICE_COMPILE__) && __has_builtin(__builtin_amdgcn_global_load_async_to_lds_b128)
#define ASYNC_LDS 1
#else
#define ASYNC_LDS 0
#endif

__device__ __forceinline__ void stage16B(const _Float16* __restrict__ src,
                                         _Float16* __restrict__ dst) {
#if ASYNC_LDS
  __builtin_amdgcn_global_load_async_to_lds_b128((gv4i_p)(src), (lv4i_p)(dst), 0, 0);
#else
  *reinterpret_cast<v8h*>(dst) = *reinterpret_cast<const v8h*>(src);
#endif
}

__device__ __forceinline__ void stage_wait() {
#if ASYNC_LDS
#if __has_builtin(__builtin_amdgcn_s_wait_asynccnt)
  __builtin_amdgcn_s_wait_asynccnt(0);
#else
  asm volatile("s_wait_asynccnt 0x0" ::: "memory");
#endif
#endif
}

// scheduling fence: keep the 12-load B-fragment clause ahead of the WMMA chain
__device__ __forceinline__ void sched_fence() {
#if defined(__HIP_DEVICE_COMPILE__) && __has_builtin(__builtin_amdgcn_sched_barrier)
  __builtin_amdgcn_sched_barrier(0);
#endif
}

// copy one 64-row tile (row stride srcld -> padded dstld), 256 threads:
// row = tid>>2, each thread moves 6 contiguous 16B chunks (no div/mod)
__device__ __forceinline__ void stage_tile(const _Float16* __restrict__ g,
                                           _Float16* __restrict__ s,
                                           int srcld, int dstld) {
  const int row = threadIdx.x >> 2;
  const int q = (threadIdx.x & 3) * 48;
  const _Float16* src = g + row * srcld + q;
  _Float16* dst = s + row * dstld + q;
#pragma unroll
  for (int j = 0; j < 6; ++j) stage16B(src + j * 8, dst + j * 8);
}

// ---- WMMA fragment loaders (CDNA5 16x16x32 f16 layouts, ISA 7.12.2) ----
// A (16x32, MxK): lane L holds row M=L&15; halves 0-7 -> K=(L>=16?8:0)+k0+i,
// halves 8-15 -> +16.  Two 16-byte chunks per lane.
// B (32x16, KxN): same pattern with "row" = N (out-feature); weights stored
// [out][in] keep K contiguous -> same loader.
__device__ __forceinline__ v16h load_frag(const _Float16* base, int ld, int row0, int k0) {
  const int lane = threadIdx.x & 31;
  const _Float16* p = base + (size_t)(row0 + (lane & 15)) * ld + k0 + ((lane >> 4) << 3);
  union { v16h v; v8h h[2]; } u;
  u.h[0] = *reinterpret_cast<const v8h*>(p);
  u.h[1] = *reinterpret_cast<const v8h*>(p + 16);
  return u.v;
}

__device__ __forceinline__ v8f wmma_f16(v16h a, v16h b, v8f c) {
  return __builtin_amdgcn_wmma_f32_16x16x32_f16(false, a, false, b, (short)0, c, false, false);
}

__device__ __forceinline__ float wave_red_sum(float v) {
  for (int m = 16; m; m >>= 1) v += __shfl_xor(v, m, 32);
  return v;
}
__device__ __forceinline__ float half_red_max(float v) {
  for (int m = 8; m; m >>= 1) v = fmaxf(v, __shfl_xor(v, m, 32));
  return v;
}
__device__ __forceinline__ float half_red_sum(float v) {
  for (int m = 8; m; m >>= 1) v += __shfl_xor(v, m, 32);
  return v;
}

// ---- fp32 -> f16 weight conversion ----
__global__ void cvt_f16_kernel(const float* __restrict__ src, _Float16* __restrict__ dst, int n) {
  int i = blockIdx.x * 256 + threadIdx.x;
  if (i < n) dst[i] = (_Float16)src[i];
}

// ---- LN1 + roll(-3,-3) + window partition + pad 49->64, NCHW fp32 -> f16 ----
__global__ void ln1_window_kernel(const float* __restrict__ x, const float* __restrict__ g,
                                  const float* __restrict__ be, _Float16* __restrict__ A) {
  const int lane = threadIdx.x & 31;
  const int task = blockIdx.x * 8 + (threadIdx.x >> 5);   // win*64 + t
  const int t = task & 63;
  const int win = task >> 6;
  _Float16* dst = A + (size_t)task * DIMC;
  if (t >= NREAL) {                                       // zero padding rows
    for (int i = 0; i < 6; ++i) dst[lane + 32 * i] = (_Float16)0.f;
    return;
  }
  const int b = win >> 6, wh = (win >> 3) & 7, ww = win & 7;
  const int h = (wh * WINSZ + t / WINSZ + 3) % IMG;       // roll by -3
  const int w = (ww * WINSZ + t % WINSZ + 3) % IMG;
  const float* px = x + (size_t)b * DIMC * HW + h * IMG + w;
  float v[6]; float s = 0.f, ss = 0.f;
#pragma unroll
  for (int i = 0; i < 6; ++i) {
    float f = px[(size_t)(lane + 32 * i) * HW];
    v[i] = f; s += f; ss += f * f;
  }
  s = wave_red_sum(s); ss = wave_red_sum(ss);
  const float mean = s * (1.f / DIMC);
  const float rstd = rsqrtf(ss * (1.f / DIMC) - mean * mean + 1e-5f);
#pragma unroll
  for (int i = 0; i < 6; ++i) {
    int c = lane + 32 * i;
    dst[c] = (_Float16)((v[i] - mean) * rstd * g[c] + be[c]);
  }
}

// ---- QKV GEMM: per window, (64x192) x (192x576) + bias -> Q,K (tok,dim), V^T (dim,tok) ----
// wave owns one 16-row strip (A frags register-resident), strides over col tiles;
// all 6 B frags preloaded per tile and pinned ahead of the WMMA chain.
__global__ void qkv_kernel(const _Float16* __restrict__ A, const _Float16* __restrict__ Wq,
                           const float* __restrict__ bq, _Float16* __restrict__ Q,
                           _Float16* __restrict__ K, _Float16* __restrict__ Vt) {
  __shared__ _Float16 sA[NTOK * 200];                     // padded ld=200 (bank-safe b128)
  const int win = blockIdx.x;
  const int tid = threadIdx.x, lane = tid & 31, wave = tid >> 5;
  const int hi = lane >> 4, nn = lane & 15;
  stage_tile(A + (size_t)win * NTOK * DIMC, sA, DIMC, 200);
  stage_wait();
  __syncthreads();

  const int rt = wave & 3;                                // row tile, 2 waves each
  const int cst = wave >> 2;                              // col start 0/1, stride 2
  v16h aFr[6];
#pragma unroll
  for (int kk = 0; kk < 6; ++kk) aFr[kk] = load_frag(sA, 200, rt * 16, kk * 32);

  int which = 0, rem = cst;                               // which = ct/12, rem = ct%12
  for (int ct = cst; ct < 36; ct += 2) {
    const int n0 = ct * 16;
    v16h bFr[6];
#pragma unroll
    for (int kk = 0; kk < 6; ++kk) bFr[kk] = load_frag(Wq, DIMC, n0, kk * 32);
    sched_fence();
    v8f acc = {};
#pragma unroll
    for (int kk = 0; kk < 6; ++kk) acc = wmma_f16(aFr[kk], bFr[kk], acc);

    const float bias = bq[n0 + nn];
    const int head = rem >> 1, d0 = (rem & 1) * 16;
    const size_t hb = (size_t)win * HEADS + head;
    if (which == 2) {
      _Float16* dp = Vt + (hb * DH + d0 + nn) * NTOK + rt * 16 + hi * 8;
#pragma unroll
      for (int r = 0; r < 8; ++r) dp[r] = (_Float16)(acc[r] + bias);
    } else {
      _Float16* dp = (which == 0 ? Q : K) + (hb * NTOK + rt * 16 + hi * 8) * DH + d0 + nn;
#pragma unroll
      for (int r = 0; r < 8; ++r) dp[(size_t)r * DH] = (_Float16)(acc[r] + bias);
    }
    rem += 2;
    if (rem >= 12) { rem -= 12; ++which; }
  }
}

// ---- attention: per (window,head): S=QK^T*s, masked softmax in-register, O=S@V ----
__global__ void attn_kernel(const _Float16* __restrict__ Q, const _Float16* __restrict__ K,
                            const _Float16* __restrict__ Vt, _Float16* __restrict__ O) {
  __shared__ _Float16 sAttn[NTOK * 72];                   // padded ld=72
  const int wgid = blockIdx.x;                            // win*HEADS + head
  const int win = wgid / HEADS, head = wgid % HEADS;
  const int lane = threadIdx.x & 31, wave = threadIdx.x >> 5;
  const int hi = lane >> 4, nn = lane & 15;
  const int m0 = wave * 16;                               // 4 waves, 16 query rows each
  const _Float16* Qh = Q + (size_t)wgid * NTOK * DH;
  const _Float16* Kh = K + (size_t)wgid * NTOK * DH;
  const _Float16* Vh = Vt + (size_t)wgid * DH * NTOK;
  const float scale = 0.17677669529663687f;               // 32^-0.5
  v8f s[4];
  v16h aq = load_frag(Qh, DH, m0, 0);                     // K=32 -> one WMMA per tile
  v16h bk[4];
#pragma unroll
  for (int tn = 0; tn < 4; ++tn) bk[tn] = load_frag(Kh, DH, tn * 16, 0);
  sched_fence();
#pragma unroll
  for (int tn = 0; tn < 4; ++tn) {
    v8f z = {};
    s[tn] = wmma_f16(aq, bk[tn], z);
  }
#pragma unroll
  for (int tn = 0; tn < 4; ++tn) {                        // scale + pad-key mask
    const bool dead = (tn * 16 + nn) >= NREAL;
#pragma unroll
    for (int r = 0; r < 8; ++r) s[tn][r] = dead ? -1e30f : s[tn][r] * scale;
  }
  // softmax directly in D layout: row m lives in one 16-lane half -> xor masks 1..8
#pragma unroll
  for (int r = 0; r < 8; ++r) {
    float mx = fmaxf(fmaxf(s[0][r], s[1][r]), fmaxf(s[2][r], s[3][r]));
    mx = half_red_max(mx);
    float sum = 0.f;
#pragma unroll
    for (int tn = 0; tn < 4; ++tn) { float e = expf(s[tn][r] - mx); s[tn][r] = e; sum += e; }
    sum = half_red_sum(sum);
    const float inv = 1.f / sum;
    const int m = m0 + r + hi * 8;
#pragma unroll
    for (int tn = 0; tn < 4; ++tn) sAttn[m * 72 + tn * 16 + nn] = (_Float16)(s[tn][r] * inv);
  }
  __syncthreads();
#pragma unroll
  for (int nt = 0; nt < 2; ++nt) {                        // O = attn(64x64) @ V(64x32)
    v16h a0 = load_frag(sAttn, 72, m0, 0);
    v16h a1 = load_frag(sAttn, 72, m0, 32);
    v16h b0 = load_frag(Vh, NTOK, nt * 16, 0);
    v16h b1 = load_frag(Vh, NTOK, nt * 16, 32);
    sched_fence();
    v8f acc = {};
    acc = wmma_f16(a0, b0, acc);
    acc = wmma_f16(a1, b1, acc);
    _Float16* dp = O + ((size_t)win * NTOK + m0 + hi * 8) * DIMC + head * DH + nt * 16 + nn;
#pragma unroll
    for (int r = 0; r < 8; ++r) dp[(size_t)r * DIMC] = (_Float16)acc[r];
  }
}

// ---- proj GEMM + bias + LN2 + window-reverse/unroll scatter to spatial f16 Y ----
__global__ void proj_ln2_kernel(const _Float16* __restrict__ O, const _Float16* __restrict__ Wp,
                                const float* __restrict__ bp, const float* __restrict__ g2,
                                const float* __restrict__ b2, _Float16* __restrict__ Y) {
  extern __shared__ char smem[];
  _Float16* sO = (_Float16*)smem;                         // 64 x 200 f16
  float* sR = (float*)(smem + NTOK * 200 * 2);            // 64 x 193 f32
  const int win = blockIdx.x;
  const int tid = threadIdx.x, lane = tid & 31, wave = tid >> 5;
  const int hi = lane >> 4, nn = lane & 15;
  stage_tile(O + (size_t)win * NTOK * DIMC, sO, DIMC, 200);
  stage_wait();
  __syncthreads();

  const int rt = wave & 3, cst = wave >> 2;
  v16h aFr[6];
#pragma unroll
  for (int kk = 0; kk < 6; ++kk) aFr[kk] = load_frag(sO, 200, rt * 16, kk * 32);
  for (int ct = cst; ct < 12; ct += 2) {
    const int n0 = ct * 16;
    v16h bFr[6];
#pragma unroll
    for (int kk = 0; kk < 6; ++kk) bFr[kk] = load_frag(Wp, DIMC, n0, kk * 32);
    sched_fence();
    v8f acc = {};
#pragma unroll
    for (int kk = 0; kk < 6; ++kk) acc = wmma_f16(aFr[kk], bFr[kk], acc);
    const float bias = bp[n0 + nn];
#pragma unroll
    for (int r = 0; r < 8; ++r) sR[(rt * 16 + r + hi * 8) * 193 + n0 + nn] = acc[r] + bias;
  }
  __syncthreads();
  const int b = win >> 6, wh = (win >> 3) & 7, ww = win & 7;
  for (int t = wave; t < NREAL; t += 8) {                 // LN2 per valid token
    float v[6]; float s = 0.f, ss = 0.f;
#pragma unroll
    for (int i = 0; i < 6; ++i) {
      float f = sR[t * 193 + lane + 32 * i];
      v[i] = f; s += f; ss += f * f;
    }
    s = wave_red_sum(s); ss = wave_red_sum(ss);
    const float mean = s * (1.f / DIMC);
    const float rstd = rsqrtf(ss * (1.f / DIMC) - mean * mean + 1e-5f);
    const int h = (wh * WINSZ + t / WINSZ + 3) % IMG;     // roll back +3
    const int w = (ww * WINSZ + t % WINSZ + 3) % IMG;
    _Float16* dst = Y + ((size_t)b * HW + h * IMG + w) * DIMC;
#pragma unroll
    for (int i = 0; i < 6; ++i) {
      int c = lane + 32 * i;
      dst[c] = (_Float16)((v[i] - mean) * rstd * g2[c] + b2[c]);
    }
  }
}

// ---- fused MLP: fc1+GELU (hidden tile in LDS) + fc2 + NCHW transpose + residual ----
__global__ void mlp_out_kernel(const _Float16* __restrict__ Y, const _Float16* __restrict__ W1,
                               const float* __restrict__ b1, const _Float16* __restrict__ W2,
                               const float* __restrict__ b2, const float* __restrict__ x,
                               float* __restrict__ out) {
  extern __shared__ char smem[];
  _Float16* sY = (_Float16*)smem;                         // 64 x 200 f16 (25600 B)
  _Float16* sH = (_Float16*)(smem + 25600);               // 64 x 776 f16 (99328 B)
  float*    sR = (float*)(smem + 25600 + 99328);          // 64 x 193 f32 (49408 B)
  const int tid = threadIdx.x, lane = tid & 31, wave = tid >> 5;
  const int hi = lane >> 4, nn = lane & 15;
  const size_t t0 = (size_t)blockIdx.x * NTOK;            // 64-token block (in-image aligned)
  stage_tile(Y + t0 * DIMC, sY, DIMC, 200);
  stage_wait();
  __syncthreads();

  const int rt = wave & 3, cst = wave >> 2;
  v16h aFr[6];
#pragma unroll
  for (int kk = 0; kk < 6; ++kk) aFr[kk] = load_frag(sY, 200, rt * 16, kk * 32);
  for (int ct = cst; ct < 48; ct += 2) {                  // fc1: 64x768
    const int n0 = ct * 16;
    v16h bFr[6];
#pragma unroll
    for (int kk = 0; kk < 6; ++kk) bFr[kk] = load_frag(W1, DIMC, n0, kk * 32);
    sched_fence();
    v8f acc = {};
#pragma unroll
    for (int kk = 0; kk < 6; ++kk) acc = wmma_f16(aFr[kk], bFr[kk], acc);
    const float bias = b1[n0 + nn];
    _Float16* hp = sH + (rt * 16 + hi * 8) * 776 + n0 + nn;
#pragma unroll
    for (int r = 0; r < 8; ++r) {
      float v = acc[r] + bias;
      v = 0.5f * v * (1.f + erff(v * 0.70710678118654752f));  // exact GELU
      hp[r * 776] = (_Float16)v;
    }
  }
  __syncthreads();
  for (int ct = cst; ct < 12; ct += 2) {                  // fc2: 64x192, K=768
    const int n0 = ct * 16;
    v8f acc = {};
#pragma unroll
    for (int g = 0; g < 4; ++g) {                         // 4 groups of 6 K-chunks
      v16h bFr[6];
#pragma unroll
      for (int j = 0; j < 6; ++j) bFr[j] = load_frag(W2, HID, n0, (g * 6 + j) * 32);
      sched_fence();
#pragma unroll
      for (int j = 0; j < 6; ++j)
        acc = wmma_f16(load_frag(sH, 776, rt * 16, (g * 6 + j) * 32), bFr[j], acc);
    }
    const float bias = b2[n0 + nn];
#pragma unroll
    for (int r = 0; r < 8; ++r) sR[(rt * 16 + r + hi * 8) * 193 + n0 + nn] = acc[r] + bias;
  }
  __syncthreads();
  const int b = (int)(t0 / HW);
  const int s0 = (int)(t0 % HW);
  for (int i = tid; i < NTOK * DIMC; i += 256) {          // coalesced NCHW store + shortcut
    const int c = i >> 6, tl = i & 63;
    const size_t gi = ((size_t)b * DIMC + c) * HW + s0 + tl;
    out[gi] = sR[tl * 193 + c] + x[gi];
  }
}

extern "C" void kernel_launch(void* const* d_in, const int* in_sizes, int n_in,
                              void* d_out, int out_size, void* d_ws, size_t ws_size,
                              hipStream_t stream) {
  (void)in_sizes; (void)n_in; (void)out_size; (void)ws_size;
  const float* x    = (const float*)d_in[0];
  const float* n1w  = (const float*)d_in[1];
  const float* n1b  = (const float*)d_in[2];
  const float* qkvw = (const float*)d_in[3];
  const float* qkvb = (const float*)d_in[4];
  const float* pw   = (const float*)d_in[5];
  const float* pb   = (const float*)d_in[6];
  const float* n2w  = (const float*)d_in[7];
  const float* n2b  = (const float*)d_in[8];
  const float* f1w  = (const float*)d_in[9];
  const float* f1b  = (const float*)d_in[10];
  const float* f2w  = (const float*)d_in[11];
  const float* f2b  = (const float*)d_in[12];
  float* out = (float*)d_out;

  char* ws = (char*)d_ws;
  size_t off = 0;
  auto alloc = [&](size_t bytes) -> void* {
    void* p = ws + off;
    off = (off + bytes + 255) & ~(size_t)255;
    return p;
  };
  _Float16* wq = (_Float16*)alloc((size_t)3 * DIMC * DIMC * 2);      // 576x192
  _Float16* wp = (_Float16*)alloc((size_t)DIMC * DIMC * 2);          // 192x192
  _Float16* w1 = (_Float16*)alloc((size_t)HID * DIMC * 2);           // 768x192
  _Float16* w2 = (_Float16*)alloc((size_t)DIMC * HID * 2);           // 192x768
  _Float16* A  = (_Float16*)alloc((size_t)NWIN * NTOK * DIMC * 2);
  _Float16* Qb = (_Float16*)alloc((size_t)NWIN * HEADS * NTOK * DH * 2);
  _Float16* Kb = (_Float16*)alloc((size_t)NWIN * HEADS * NTOK * DH * 2);
  _Float16* Vt = (_Float16*)alloc((size_t)NWIN * HEADS * DH * NTOK * 2);
  _Float16* Oa = (_Float16*)alloc((size_t)NWIN * NTOK * DIMC * 2);
  _Float16* Yb = (_Float16*)alloc((size_t)16 * HW * DIMC * 2);

  cvt_f16_kernel<<<(3 * DIMC * DIMC + 255) / 256, 256, 0, stream>>>(qkvw, wq, 3 * DIMC * DIMC);
  cvt_f16_kernel<<<(DIMC * DIMC + 255) / 256, 256, 0, stream>>>(pw, wp, DIMC * DIMC);
  cvt_f16_kernel<<<(HID * DIMC + 255) / 256, 256, 0, stream>>>(f1w, w1, HID * DIMC);
  cvt_f16_kernel<<<(DIMC * HID + 255) / 256, 256, 0, stream>>>(f2w, w2, DIMC * HID);

  ln1_window_kernel<<<NWIN * NTOK / 8, 256, 0, stream>>>(x, n1w, n1b, A);
  qkv_kernel<<<NWIN, 256, 0, stream>>>(A, wq, qkvb, Qb, Kb, Vt);
  attn_kernel<<<NWIN * HEADS, 128, 0, stream>>>(Qb, Kb, Vt, Oa);
  proj_ln2_kernel<<<NWIN, 256, NTOK * 200 * 2 + NTOK * 193 * 4, stream>>>(Oa, wp, pb, n2w, n2b, Yb);
  mlp_out_kernel<<<16 * HW / NTOK, 256, 25600 + 99328 + NTOK * 193 * 4, stream>>>(
      Yb, w1, f1b, w2, f2b, x, out);
}
